// PlainRNN_24378234372574
// MI455X (gfx1250) — matrix operand order repaired and checked by reading, not compiled, not run on hardware
//
#include <hip/hip_runtime.h>
#include <hip/hip_bf16.h>
#include <math.h>

// ---------------------------------------------------------------------------
// PlainRNN on gfx1250: persistent cooperative kernel, bf16 WMMA, f32 accum.
// Round 2: branch-free fast tanh, A-fragment reuse (2 N-tiles per wave).
// ---------------------------------------------------------------------------
typedef __bf16 bf16;
typedef __attribute__((ext_vector_type(8)))  __bf16 v8bf;
typedef __attribute__((ext_vector_type(16))) __bf16 v16bf;
typedef __attribute__((ext_vector_type(8)))  float  v8f;

#define INPUT   128
#define HIDDEN  1024
#define OUTPUT  128
#define BATCH   256
#define TSTEPS  512
#define KTOT    (INPUT + HIDDEN)     // 1152
#define GRID    64
#define TPB     256                  // 8 waves (wave32)
#define NWAVES  (GRID * (TPB / 32))  // 512 waves

union ABfrag { v16bf v16; v8bf v8[2]; };

// Branch-free tanh: clamp, then 1 - 2/(e^{2x}+1). v_exp_f32 + v_rcp_f32,
// no EXEC divergence (unlike libm tanhf).
__device__ __forceinline__ float fast_tanh(float x) {
    float xc = fminf(fmaxf(x, -9.0f), 9.0f);
    float e2 = __expf(2.0f * xc);
    float r  = __builtin_amdgcn_rcpf(e2 + 1.0f);
    return 1.0f - 2.0f * r;
}

// ---------------------------------------------------------------------------
// f32 -> bf16 conversion (grid-stride)
// ---------------------------------------------------------------------------
__global__ void f32_to_bf16_kernel(const float* __restrict__ src,
                                   bf16* __restrict__ dst, int n) {
    for (int i = blockIdx.x * blockDim.x + threadIdx.x; i < n;
         i += gridDim.x * blockDim.x)
        dst[i] = (bf16)src[i];
}

// ---------------------------------------------------------------------------
// Init: xh buffer 0 = [bf16(x0) | 0], combined bias, out[0] = x0, counter = 0
// ---------------------------------------------------------------------------
__global__ void rnn_init_kernel(const float* __restrict__ x0,
                                const float* __restrict__ b_i2h,
                                const float* __restrict__ b_h2h,
                                bf16* __restrict__ xh0,
                                float* __restrict__ bias_comb,
                                float* __restrict__ out,
                                unsigned* __restrict__ cnt) {
    const int n = BATCH * KTOT;
    for (int i = blockIdx.x * blockDim.x + threadIdx.x; i < n;
         i += gridDim.x * blockDim.x) {
        int r = i / KTOT, c = i % KTOT;
        xh0[i] = (c < INPUT) ? (bf16)x0[r * INPUT + c] : (bf16)0.0f;
        if (i < BATCH * INPUT) out[i] = x0[i];          // trajectory[0] = x_0
        if (i < HIDDEN)        bias_comb[i] = b_i2h[i] + b_h2h[i];
        if (i == 0)            *cnt = 0u;
    }
}

// ---------------------------------------------------------------------------
// Grid-wide barrier: monotonic counter, agent scope.
// ---------------------------------------------------------------------------
__device__ __forceinline__ void grid_barrier(unsigned* cnt, unsigned expected) {
    __threadfence();            // release our global writes to device scope
    __syncthreads();
    if (threadIdx.x == 0) {
        __hip_atomic_fetch_add(cnt, 1u, __ATOMIC_RELEASE,
                               __HIP_MEMORY_SCOPE_AGENT);
        while (__hip_atomic_load(cnt, __ATOMIC_ACQUIRE,
                                 __HIP_MEMORY_SCOPE_AGENT) < expected)
            __builtin_amdgcn_s_sleep(1);
    }
    __syncthreads();
}

// ---------------------------------------------------------------------------
// Persistent RNN kernel: 511 steps, 2 phases/step, WMMA bf16 16x16x32.
// Each wave: one A stream feeding two N-tiles (independent accum chains).
// ---------------------------------------------------------------------------
__global__ void __launch_bounds__(TPB)
rnn_persistent_kernel(bf16* __restrict__ xh0, bf16* __restrict__ xh1,
                      const bf16* __restrict__ Wc,    // [KTOT x HIDDEN]
                      const bf16* __restrict__ Wo,    // [HIDDEN x OUTPUT]
                      const float* __restrict__ bias_h,  // combined, [HIDDEN]
                      const float* __restrict__ bias_o,  // [OUTPUT]
                      float* __restrict__ out,           // [T, BATCH, OUTPUT]
                      unsigned* __restrict__ cnt) {
    const int lane = threadIdx.x & 31;
    const int wave = threadIdx.x >> 5;
    const int wgid = blockIdx.x * (TPB / 32) + wave;   // 0..511
    const int mrow = lane & 15;                         // A row / D col
    const int half = lane >> 4;                         // K-split half

    // Phase H tile assignment: 16 M-tiles x 64 N-tiles = 1024 tiles.
    // Wave w: M-tile = w>>5, N-tiles = (w&31) and (w&31)+32  (shared A).
    const int h_m0  = (wgid >> 5) << 4;
    const int h_n0a = (wgid & 31) << 4;
    const int h_n0b = h_n0a + 512;

    // Phase O tile assignment: 16 M-tiles x 8 N-tiles = 128 tiles on waves 0..63.
    // Wave w: M-tile = w>>2, N-tiles = 2*(w&3) and 2*(w&3)+1 (shared A).
    const int o_m0  = (wgid >> 2) << 4;
    const int o_n0a = (wgid & 3) << 5;
    const int o_n0b = o_n0a + 16;

    unsigned nbar = 0;

    for (int t = 0; t < TSTEPS - 1; ++t) {
        const bf16* cur = (t & 1) ? xh1 : xh0;
        bf16*       nxt = (t & 1) ? xh0 : xh1;

        // ---------------- Phase H: h_new = tanh([x|h] @ Wc + bias_h) --------
        {
            v8f c0, c1;
            const float bi0 = bias_h[h_n0a + mrow];
            const float bi1 = bias_h[h_n0b + mrow];
            #pragma unroll
            for (int i = 0; i < 8; ++i) { c0[i] = bi0; c1[i] = bi1; }

            const bf16* arow  = cur + (h_m0 + mrow) * KTOT + 8 * half;
            const bf16* brow0 = Wc + (8 * half) * HIDDEN + h_n0a + mrow;
            const bf16* brow1 = brow0 + 512;

            #pragma unroll 2
            for (int k0 = 0; k0 < KTOT; k0 += 32) {
                ABfrag a, b0, b1;
                a.v8[0] = *(const v8bf*)(arow + k0);
                a.v8[1] = *(const v8bf*)(arow + k0 + 16);
                #pragma unroll
                for (int s = 0; s < 16; ++s) {
                    const size_t kk = (size_t)(k0 + (s & 7) + ((s >> 3) << 4));
                    b0.v16[s] = brow0[kk * HIDDEN];
                    b1.v16[s] = brow1[kk * HIDDEN];
                }
                c0 = __builtin_amdgcn_wmma_f32_16x16x32_bf16(
                         false, a.v16, false, b0.v16, (short)0, c0, false, false);
                c1 = __builtin_amdgcn_wmma_f32_16x16x32_bf16(
                         false, a.v16, false, b1.v16, (short)0, c1, false, false);
            }

            // tanh + store bf16 into h-columns of next activation buffer
            bf16* dcol0 = nxt + INPUT + h_n0a + mrow;
            bf16* dcol1 = nxt + INPUT + h_n0b + mrow;
            #pragma unroll
            for (int r = 0; r < 8; ++r) {
                const size_t m = (size_t)(h_m0 + r + 8 * half);
                dcol0[m * KTOT] = (bf16)fast_tanh(c0[r]);
                dcol1[m * KTOT] = (bf16)fast_tanh(c1[r]);
            }
        }

        ++nbar; grid_barrier(cnt, nbar * GRID);

        // ---------------- Phase O: x_new = h_new @ Wo + bias_o --------------
        if (wgid < 64) {
            v8f c0, c1;
            const float bi0 = bias_o[o_n0a + mrow];
            const float bi1 = bias_o[o_n0b + mrow];
            #pragma unroll
            for (int i = 0; i < 8; ++i) { c0[i] = bi0; c1[i] = bi1; }

            const bf16* arow  = nxt + INPUT + (o_m0 + mrow) * KTOT + 8 * half;
            const bf16* brow0 = Wo + (8 * half) * OUTPUT + o_n0a + mrow;
            const bf16* brow1 = brow0 + 16;

            #pragma unroll 2
            for (int k0 = 0; k0 < HIDDEN; k0 += 32) {
                ABfrag a, b0, b1;
                a.v8[0] = *(const v8bf*)(arow + k0);
                a.v8[1] = *(const v8bf*)(arow + k0 + 16);
                #pragma unroll
                for (int s = 0; s < 16; ++s) {
                    const size_t kk = (size_t)(k0 + (s & 7) + ((s >> 3) << 4));
                    b0.v16[s] = brow0[kk * OUTPUT];
                    b1.v16[s] = brow1[kk * OUTPUT];
                }
                c0 = __builtin_amdgcn_wmma_f32_16x16x32_bf16(
                         false, a.v16, false, b0.v16, (short)0, c0, false, false);
                c1 = __builtin_amdgcn_wmma_f32_16x16x32_bf16(
                         false, a.v16, false, b1.v16, (short)0, c1, false, false);
            }

            float* ocol0 = out + (size_t)(t + 1) * BATCH * OUTPUT + o_n0a + mrow;
            float* ocol1 = ocol0 + 16;
            bf16*  xcol0 = nxt + o_n0a + mrow;
            bf16*  xcol1 = xcol0 + 16;
            #pragma unroll
            for (int r = 0; r < 8; ++r) {
                const size_t m = (size_t)(o_m0 + r + 8 * half);
                ocol0[m * OUTPUT] = c0[r];          // trajectory output
                ocol1[m * OUTPUT] = c1[r];
                xcol0[m * KTOT]   = (bf16)c0[r];    // feed back as next x
                xcol1[m * KTOT]   = (bf16)c1[r];
            }
        }

        ++nbar; grid_barrier(cnt, nbar * GRID);
    }
}

// ---------------------------------------------------------------------------
// Host side
// ---------------------------------------------------------------------------
extern "C" void kernel_launch(void* const* d_in, const int* in_sizes, int n_in,
                              void* d_out, int out_size, void* d_ws, size_t ws_size,
                              hipStream_t stream) {
    const float* x0    = (const float*)d_in[0];
    const float* W_i2h = (const float*)d_in[1];
    const float* b_i2h = (const float*)d_in[2];
    const float* W_h2h = (const float*)d_in[3];
    const float* b_h2h = (const float*)d_in[4];
    const float* W_h2o = (const float*)d_in[5];
    const float* b_h2o = (const float*)d_in[6];
    float* out = (float*)d_out;

    // Workspace carve-up
    bf16* Wc   = (bf16*)d_ws;                 // [1152 x 1024] bf16
    bf16* Wo   = Wc  + (size_t)KTOT * HIDDEN; // [1024 x 128]  bf16
    bf16* xh0  = Wo  + (size_t)HIDDEN * OUTPUT;
    bf16* xh1  = xh0 + (size_t)BATCH * KTOT;
    float* bias = (float*)(xh1 + (size_t)BATCH * KTOT); // [1024] f32
    unsigned* cnt = (unsigned*)(bias + HIDDEN);

    // Convert weights to bf16 (Wc = [W_i2h ; W_h2h] stacked along K)
    {
        int n1 = INPUT * HIDDEN;
        f32_to_bf16_kernel<<<(n1 + TPB - 1) / TPB, TPB, 0, stream>>>(W_i2h, Wc, n1);
        int n2 = HIDDEN * HIDDEN;
        f32_to_bf16_kernel<<<(n2 + TPB - 1) / TPB, TPB, 0, stream>>>(
            W_h2h, Wc + (size_t)INPUT * HIDDEN, n2);
        int n3 = HIDDEN * OUTPUT;
        f32_to_bf16_kernel<<<(n3 + TPB - 1) / TPB, TPB, 0, stream>>>(W_h2o, Wo, n3);
    }

    // Init activations, combined bias, out[0], barrier counter
    {
        int n = BATCH * KTOT;
        rnn_init_kernel<<<(n + TPB - 1) / TPB, TPB, 0, stream>>>(
            x0, b_i2h, b_h2h, xh0, bias, out, cnt);
    }

    // Persistent RNN loop
    rnn_persistent_kernel<<<GRID, TPB, 0, stream>>>(
        xh0, xh1, Wc, Wo, bias, b_h2o, out, cnt);
}
